// ProbAttentionLayer_33243046871128
// MI455X (gfx1250) — matrix operand order repaired and checked by reading, not compiled, not run on hardware
//
#include <hip/hip_runtime.h>
#include <hip/hip_bf16.h>
#include <math.h>

typedef __attribute__((ext_vector_type(16))) _Float16 v16h;
typedef __attribute__((ext_vector_type(8)))  float    v8f;
typedef __attribute__((ext_vector_type(4)))  int      v4i;

#define B_    4
#define L_    2048
#define D_    1024
#define H_    16
#define HD_   64
#define MTOT  (B_ * L_)   // 8192

// ---------------------------------------------------------------------------
// CDNA5 async global->LDS copy (ASYNCcnt-tracked), with sync fallback
// ---------------------------------------------------------------------------
#if defined(__gfx1250__) && __has_builtin(__builtin_amdgcn_global_load_async_to_lds_b128)
#define ASYNC_LDS 1
#else
#define ASYNC_LDS 0
#endif

#if ASYNC_LDS
__device__ __forceinline__ void async_copy_b128(const void* g, void* l) {
  __builtin_amdgcn_global_load_async_to_lds_b128(
      (__attribute__((address_space(1))) v4i*)(unsigned long long)(size_t)g,
      (__attribute__((address_space(3))) v4i*)l, 0, 0);
}
#endif

__device__ __forceinline__ void wait_async0() {
#if ASYNC_LDS
#if __has_builtin(__builtin_amdgcn_s_wait_asynccnt)
  __builtin_amdgcn_s_wait_asynccnt(0);
#else
  asm volatile("s_wait_asynccnt 0" ::: "memory");
#endif
#endif
}

// ---------------------------------------------------------------------------
// WMMA helpers (CDNA5 wave32, v_wmma_f32_16x16x32_f16)
// ---------------------------------------------------------------------------
__device__ __forceinline__ v8f wmma_f16(v16h a, v16h b, v8f c) {
  return __builtin_amdgcn_wmma_f32_16x16x32_f16(false, a, false, b, (short)0, c,
                                                false, false);
}

// 16-bit A matrix 16x32 (MxK): lane m = lane&15; halves h:
//   k = (h%8) + (h/8)*16 + (lane>=16 ? 8 : 0)
__device__ __forceinline__ v16h load_a(const _Float16* row, int k0, int lane) {
  union { v16h h; unsigned u[8]; } r;
  const int base = k0 + ((lane & 16) ? 8 : 0);
#pragma unroll
  for (int j = 0; j < 4; ++j) {
    r.u[j]     = *(const unsigned*)(row + base + 2 * j);
    r.u[j + 4] = *(const unsigned*)(row + base + 16 + 2 * j);
  }
  return r.h;
}

// 16-bit B matrix 32x16 (KxN): lane n = lane&15;
//   k = (h%8) + (h/8)*16 + (lane>=16 ? 8 : 0)
__device__ __forceinline__ v16h load_b(const _Float16* rowN, int k0, int lane) {
  union { v16h h; unsigned u[8]; } r;
  const int base = k0 + ((lane & 16) ? 8 : 0);
#pragma unroll
  for (int j = 0; j < 4; ++j) {
    r.u[j]     = *(const unsigned*)(rowN + base + 2 * j);
    r.u[j + 4] = *(const unsigned*)(rowN + base + 16 + 2 * j);
  }
  return r.h;
}

__device__ __forceinline__ float rmax16(float v) {
#pragma unroll
  for (int m = 1; m < 16; m <<= 1) v = fmaxf(v, __shfl_xor(v, m, 32));
  return v;
}
__device__ __forceinline__ float rsum16(float v) {
#pragma unroll
  for (int m = 1; m < 16; m <<= 1) v += __shfl_xor(v, m, 32);
  return v;
}

// ---------------------------------------------------------------------------
// Conversion kernels
// ---------------------------------------------------------------------------
__global__ void cvt_f32_to_f16(const float* __restrict__ src,
                               _Float16* __restrict__ dst, int n) {
  for (int i = blockIdx.x * blockDim.x + threadIdx.x; i < n;
       i += gridDim.x * blockDim.x)
    dst[i] = (_Float16)src[i];
}

__global__ void cvt_transpose_w(const float* __restrict__ w,
                                _Float16* __restrict__ wT) {
  const int i = blockIdx.x * blockDim.x + threadIdx.x;  // 0 .. D*D-1
  const int r = i >> 10, c = i & 1023;
  wT[(size_t)c * D_ + r] = (_Float16)w[i];
}

// ---------------------------------------------------------------------------
// QKV projection GEMM. grid = (MTOT/16, 3), block = 256 (8 waves).
// Q,K stored [B,H,L,64] (Q pre-scaled by 1/8); V stored transposed [B,H,64,L].
// ---------------------------------------------------------------------------
__global__ __launch_bounds__(256) void qkv_gemm(
    const _Float16* __restrict__ xh, const _Float16* __restrict__ wT3,
    const float* __restrict__ bq, const float* __restrict__ bk,
    const float* __restrict__ bv, _Float16* __restrict__ qkv) {
  const int lane = threadIdx.x & 31, wave = threadIdx.x >> 5;
  const int n = lane & 15, g = lane >> 4;
  const int which = blockIdx.y;
  const _Float16* wT = wT3 + (size_t)which * D_ * D_;
  const float* bias = (which == 0) ? bq : (which == 1) ? bk : bv;
  _Float16* outm = qkv + (size_t)which * MTOT * D_;
  const int gm0 = blockIdx.x * 16;
  const int n0 = wave * 128;
  const _Float16* arow = xh + (size_t)(gm0 + n) * D_;

  v8f zero = {0.f, 0.f, 0.f, 0.f, 0.f, 0.f, 0.f, 0.f};
  v8f c[8];
#pragma unroll
  for (int t = 0; t < 8; ++t) c[t] = zero;

  for (int kk = 0; kk < D_; kk += 32) {
    v16h a = load_a(arow, kk, lane);
#pragma unroll
    for (int t = 0; t < 8; ++t) {
      v16h b = load_b(wT + (size_t)(n0 + t * 16 + n) * D_, kk, lane);
      c[t] = wmma_f16(a, b, c[t]);
    }
  }

  const float scale = (which == 0) ? 0.125f : 1.0f;
#pragma unroll
  for (int t = 0; t < 8; ++t) {
    const int col = n0 + t * 16 + n;
    const float bcol = bias[col];
    const int h = col >> 6, hd = col & 63;
#pragma unroll
    for (int r = 0; r < 8; ++r) {
      const int gm = gm0 + r + 8 * g;
      const int bb = gm >> 11, l = gm & (L_ - 1);
      const _Float16 val = (_Float16)((c[t][r] + bcol) * scale);
      if (which == 2)  // V transposed: [bh][hd][l]
        outm[(((size_t)bb * H_ + h) * HD_ + hd) * L_ + l] = val;
      else             // Q,K: [bh][l][hd]
        outm[(((size_t)bb * H_ + h) * L_ + l) * HD_ + hd] = val;
    }
  }
}

// ---------------------------------------------------------------------------
// K/V tile stage-in (double buffered). K tile: [32 keys][64 hd] contiguous
// 4KB; V tile: 64 rows (hd) x 64B from the pre-transposed V. In the async
// path wave0 issues the K copy, wave1 the V copy (8 x b128 each).
// ---------------------------------------------------------------------------
__device__ __forceinline__ void load_kv_tiles(
    const _Float16* __restrict__ kbase, const _Float16* __restrict__ vbase,
    int kt, _Float16 (*Kt)[64], _Float16 (*Vt)[32], int wave, int lane,
    int tid) {
#if ASYNC_LDS
  if (wave == 0) {
    const char* src = (const char*)(kbase + (size_t)kt * HD_);
    char* dst = (char*)&Kt[0][0];
#pragma unroll
    for (int j = 0; j < 8; ++j)
      async_copy_b128(src + j * 512 + lane * 16, dst + j * 512 + lane * 16);
  } else if (wave == 1) {
    const int row = lane >> 2, ch = (lane & 3) * 16;
#pragma unroll
    for (int j = 0; j < 8; ++j) {
      const char* src =
          (const char*)(vbase + (size_t)(j * 8 + row) * L_ + kt) + ch;
      char* dst = (char*)&Vt[j * 8 + row][0] + ch;
      async_copy_b128(src, dst);
    }
  }
#else
  const unsigned* kg = (const unsigned*)kbase;
  const unsigned* vg = (const unsigned*)vbase;
  unsigned* kd = (unsigned*)&Kt[0][0];
  unsigned* vd = (unsigned*)&Vt[0][0];
#pragma unroll
  for (int i = 0; i < 4; ++i) {
    const int idx = tid + i * 256;          // 0..1023 dwords
    kd[idx] = kg[kt * 32 + idx];
    const int row = idx >> 4, c = idx & 15; // V: 16 dwords per hd row
    vd[idx] = vg[row * 1024 + (kt >> 1) + c];
  }
#endif
}

// ---------------------------------------------------------------------------
// Flash attention: grid = (B*H, L/128), block = 256. Wave handles 16 q rows.
// Double-buffered async K/V tiles, online softmax, all matmuls on WMMA.
// ---------------------------------------------------------------------------
__global__ __launch_bounds__(256) void flash_attn(
    const _Float16* __restrict__ q, const _Float16* __restrict__ k,
    const _Float16* __restrict__ vT, _Float16* __restrict__ att) {
  __shared__ __align__(16) _Float16 Kt[2][32][64];   // [buf][key][hd]
  __shared__ __align__(16) _Float16 Vt[2][64][32];   // [buf][hd][key]
  __shared__ __align__(16) _Float16 Pb[8][16][32];   // per-wave P bounce

  const int lane = threadIdx.x & 31, wave = threadIdx.x >> 5;
  const int tid = threadIdx.x;
  const int n = lane & 15, g = lane >> 4;
  const int bh = blockIdx.x;
  const int q0 = blockIdx.y * 128 + wave * 16;

  const _Float16* kbase = k + (size_t)bh * L_ * HD_;   // [key][hd]
  const _Float16* vbase = vT + (size_t)bh * HD_ * L_;  // [hd][key]

  const _Float16* qrow = q + ((size_t)bh * L_ + (q0 + n)) * HD_;
  v16h qa0 = load_a(qrow, 0, lane);
  v16h qa1 = load_a(qrow, 32, lane);

  v8f zero = {0.f, 0.f, 0.f, 0.f, 0.f, 0.f, 0.f, 0.f};
  float mrun[8], lrun[8];
  v8f o[4];
#pragma unroll
  for (int r = 0; r < 8; ++r) { mrun[r] = -3.0e38f; lrun[r] = 0.f; }
#pragma unroll
  for (int t = 0; t < 4; ++t) o[t] = zero;

  load_kv_tiles(kbase, vbase, 0, Kt[0], Vt[0], wave, lane, tid);

  const int NT = L_ / 32;
  for (int it = 0; it < NT; ++it) {
    const int buf = it & 1;
    wait_async0();
    __syncthreads();
    if (it + 1 < NT)
      load_kv_tiles(kbase, vbase, (it + 1) * 32, Kt[buf ^ 1], Vt[buf ^ 1],
                    wave, lane, tid);

    // scores S[16x32] = Q[16x64] * K_tile^T
    v8f c0 = zero, c1 = zero;
    {
      v16h b00 = load_b(&Kt[buf][n][0], 0, lane);
      v16h b01 = load_b(&Kt[buf][n][0], 32, lane);
      c0 = wmma_f16(qa0, b00, c0);
      c0 = wmma_f16(qa1, b01, c0);
      v16h b10 = load_b(&Kt[buf][16 + n][0], 0, lane);
      v16h b11 = load_b(&Kt[buf][16 + n][0], 32, lane);
      c1 = wmma_f16(qa0, b10, c1);
      c1 = wmma_f16(qa1, b11, c1);
    }

    // online softmax
    float corr[8];
#pragma unroll
    for (int r = 0; r < 8; ++r) {
      const float rowm = rmax16(fmaxf(c0[r], c1[r]));
      const float mnew = fmaxf(mrun[r], rowm);
      corr[r] = __expf(mrun[r] - mnew);
      c0[r] = __expf(c0[r] - mnew);
      c1[r] = __expf(c1[r] - mnew);
      lrun[r] = lrun[r] * corr[r] + rsum16(c0[r] + c1[r]);
      mrun[r] = mnew;
    }
#pragma unroll
    for (int t = 0; t < 4; ++t)
#pragma unroll
      for (int r = 0; r < 8; ++r) o[t][r] *= corr[r];

    // P (C layout) -> per-wave LDS -> A layout fragment
#pragma unroll
    for (int r = 0; r < 8; ++r) {
      const int m = r + 8 * g;
      Pb[wave][m][n] = (_Float16)c0[r];
      Pb[wave][m][16 + n] = (_Float16)c1[r];
    }
    asm volatile("s_wait_dscnt 0" ::: "memory");
    v16h pa = load_a(&Pb[wave][n][0], 0, lane);

    // O[16x64] += P[16x32] * V_tile
#pragma unroll
    for (int t = 0; t < 4; ++t) {
      v16h bv_ = load_b(&Vt[buf][t * 16 + n][0], 0, lane);
      o[t] = wmma_f16(pa, bv_, o[t]);
    }
    __syncthreads();
  }

  const int bb = bh >> 4, h = bh & 15;
#pragma unroll
  for (int r = 0; r < 8; ++r) {
    const float inv = 1.0f / lrun[r];
    const int row = q0 + r + 8 * g;
#pragma unroll
    for (int t = 0; t < 4; ++t) {
      const int col = h * HD_ + t * 16 + n;
      att[((size_t)bb * L_ + row) * D_ + col] = (_Float16)(o[t][r] * inv);
    }
  }
}

// ---------------------------------------------------------------------------
// Output projection + bias + residual + LayerNorm. grid = MTOT/16, block 256.
// ---------------------------------------------------------------------------
__global__ __launch_bounds__(256) void outproj_ln(
    const _Float16* __restrict__ att, const _Float16* __restrict__ woT,
    const float* __restrict__ bo, const float* __restrict__ x,
    const float* __restrict__ gamma, const float* __restrict__ beta,
    float* __restrict__ out) {
  __shared__ float red[16][8][2];
  __shared__ float stats[16][2];
  const int lane = threadIdx.x & 31, wave = threadIdx.x >> 5;
  const int n = lane & 15, g = lane >> 4;
  const int gm0 = blockIdx.x * 16;
  const int n0 = wave * 128;
  const _Float16* arow = att + (size_t)(gm0 + n) * D_;

  v8f zero = {0.f, 0.f, 0.f, 0.f, 0.f, 0.f, 0.f, 0.f};
  v8f c[8];
#pragma unroll
  for (int t = 0; t < 8; ++t) c[t] = zero;

  for (int kk = 0; kk < D_; kk += 32) {
    v16h a = load_a(arow, kk, lane);
#pragma unroll
    for (int t = 0; t < 8; ++t) {
      v16h b = load_b(woT + (size_t)(n0 + t * 16 + n) * D_, kk, lane);
      c[t] = wmma_f16(a, b, c[t]);
    }
  }

  float ps[8] = {0.f, 0.f, 0.f, 0.f, 0.f, 0.f, 0.f, 0.f};
  float ps2[8] = {0.f, 0.f, 0.f, 0.f, 0.f, 0.f, 0.f, 0.f};
#pragma unroll
  for (int t = 0; t < 8; ++t) {
    const int col = n0 + t * 16 + n;
    const float bcol = bo[col];
#pragma unroll
    for (int r = 0; r < 8; ++r) {
      const int gm = gm0 + r + 8 * g;
      const float y = c[t][r] + bcol + x[(size_t)gm * D_ + col];
      c[t][r] = y;
      ps[r] += y;
      ps2[r] += y * y;
    }
  }
#pragma unroll
  for (int r = 0; r < 8; ++r) { ps[r] = rsum16(ps[r]); ps2[r] = rsum16(ps2[r]); }
  if (n == 0) {
#pragma unroll
    for (int r = 0; r < 8; ++r) {
      red[r + 8 * g][wave][0] = ps[r];
      red[r + 8 * g][wave][1] = ps2[r];
    }
  }
  __syncthreads();
  if (threadIdx.x < 16) {
    float s = 0.f, s2 = 0.f;
#pragma unroll
    for (int w = 0; w < 8; ++w) {
      s += red[threadIdx.x][w][0];
      s2 += red[threadIdx.x][w][1];
    }
    const float mu = s * (1.0f / D_);
    const float var = s2 * (1.0f / D_) - mu * mu;
    stats[threadIdx.x][0] = mu;
    stats[threadIdx.x][1] = rsqrtf(var + 1e-5f);
  }
  __syncthreads();
#pragma unroll
  for (int t = 0; t < 8; ++t) {
    const int col = n0 + t * 16 + n;
    const float ga = gamma[col], be = beta[col];
#pragma unroll
    for (int r = 0; r < 8; ++r) {
      const int m = r + 8 * g;
      out[(size_t)(gm0 + m) * D_ + col] =
          (c[t][r] - stats[m][0]) * stats[m][1] * ga + be;
    }
  }
}

// ---------------------------------------------------------------------------
extern "C" void kernel_launch(void* const* d_in, const int* in_sizes, int n_in,
                              void* d_out, int out_size, void* d_ws,
                              size_t ws_size, hipStream_t stream) {
  const float* x = (const float*)d_in[0];
  const float* Wq = (const float*)d_in[1];
  const float* bq = (const float*)d_in[2];
  const float* Wk = (const float*)d_in[3];
  const float* bk = (const float*)d_in[4];
  const float* Wv = (const float*)d_in[5];
  const float* bv = (const float*)d_in[6];
  const float* Wo = (const float*)d_in[7];
  const float* bo = (const float*)d_in[8];
  const float* gamma = (const float*)d_in[9];
  const float* beta = (const float*)d_in[10];
  float* out = (float*)d_out;

  char* ws = (char*)d_ws;
  _Float16* xh = (_Float16*)(ws);                    // 16 MB
  _Float16* wT = (_Float16*)(ws + (16ull << 20));    // 3 x 2 MB (q,k,v)
  _Float16* woT = wT + (size_t)3 * D_ * D_;          // 2 MB
  _Float16* qkv = (_Float16*)(ws + (24ull << 20));   // 48 MB (q, k, vT)
  _Float16* attb = (_Float16*)(ws + (72ull << 20));  // 16 MB
  (void)in_sizes; (void)n_in; (void)out_size; (void)ws_size;

  cvt_f32_to_f16<<<4096, 256, 0, stream>>>(x, xh, MTOT * D_);
  cvt_transpose_w<<<4096, 256, 0, stream>>>(Wq, wT);
  cvt_transpose_w<<<4096, 256, 0, stream>>>(Wk, wT + (size_t)D_ * D_);
  cvt_transpose_w<<<4096, 256, 0, stream>>>(Wv, wT + (size_t)2 * D_ * D_);
  cvt_transpose_w<<<4096, 256, 0, stream>>>(Wo, woT);

  qkv_gemm<<<dim3(MTOT / 16, 3), 256, 0, stream>>>(xh, wT, bq, bk, bv, qkv);

  flash_attn<<<dim3(B_ * H_, L_ / 128), 256, 0, stream>>>(
      qkv, qkv + (size_t)MTOT * D_, qkv + (size_t)2 * MTOT * D_, attb);

  outproj_ln<<<MTOT / 16, 256, 0, stream>>>(attb, woT, bo, x, gamma, beta, out);
}
